// MagicMemory_12850542150219
// MI455X (gfx1250) — matrix operand order repaired and checked by reading, not compiled, not run on hardware
//
#include <hip/hip_runtime.h>

typedef __attribute__((ext_vector_type(16))) __bf16   v16bf;
typedef __attribute__((ext_vector_type(8)))  float    v8f;
typedef __attribute__((ext_vector_type(8)))  unsigned v8u;

#define N_KEYS  65536
#define D_TOT   1024
#define DXX     768
#define DYY     256
#define BQ      4096
#define THRESHV 0.01f

#define TB      64                 // queries per workgroup
#define TN      128                // keys per inner pass
#define KS      32                 // K-slice per WMMA step
#define KSTEPS  (D_TOT / KS)       // 32
#define KP      (KS / 2)           // 16 packed bf16-pair words per row
#define LDP     20                 // padded LDS row stride (u32): conflict-free, 16B-aligned
#define NSPLIT  64                 // key chunks (grid.y)
#define CHUNK   (N_KEYS / NSPLIT)  // 1024 keys per chunk
#define PASSES  (CHUNK / TN)       // 8
#define PARTS   (NSPLIT * 2)       // partial slots per query (2 n-groups/WG)

// ---------------- Kernel 1: exact f32 key norms (one streaming pass) -------
__global__ __launch_bounds__(256) void knorm_kernel(const float* __restrict__ keys,
                                                    float* __restrict__ knorm) {
  int n = blockIdx.x * 256 + threadIdx.x;
  const float4* row = (const float4*)(keys + (size_t)n * D_TOT);
  float s = 0.f;
#pragma unroll 8
  for (int i = 0; i < D_TOT / 4; ++i) {
    float4 v = row[i];
    s += v.x * v.x + v.y * v.y + v.z * v.z + v.w * v.w;
  }
  knorm[n] = s;
}

// pack 4 floats into 2 hi-plane words + 2 lo-plane words (bf16 hi/lo split)
__device__ __forceinline__ uint4 cvt_hilo(float4 v) {
  __bf16 h0 = (__bf16)v.x, h1 = (__bf16)v.y, h2 = (__bf16)v.z, h3 = (__bf16)v.w;
  float  l0 = v.x - (float)h0, l1 = v.y - (float)h1;
  float  l2 = v.z - (float)h2, l3 = v.w - (float)h3;
  union { __bf16 h[2]; unsigned u; } a, b, c, d;
  a.h[0] = h0;         a.h[1] = h1;
  b.h[0] = h2;         b.h[1] = h3;
  c.h[0] = (__bf16)l0; c.h[1] = (__bf16)l1;
  d.h[0] = (__bf16)l2; d.h[1] = (__bf16)l3;
  uint4 r; r.x = a.u; r.y = b.u; r.z = c.u; r.w = d.u;
  return r;
}

union FragCast { v8u u; v16bf b; };

// ------- Kernel 2: WMMA bf16-split distances + running argmin per chunk ----
__global__ __launch_bounds__(256) void dist_argmin_kernel(
    const float* __restrict__ keys, const float* __restrict__ knorm,
    const float* __restrict__ X, const float* __restrict__ Y,
    float* __restrict__ pmin, int* __restrict__ pidx) {
  __shared__ __align__(16) unsigned As_hi[TB][LDP];
  __shared__ __align__(16) unsigned As_lo[TB][LDP];
  __shared__ __align__(16) unsigned Bs_hi[TN][LDP];
  __shared__ __align__(16) unsigned Bs_lo[TN][LDP];

  const int tid  = threadIdx.x;
  const int lane = tid & 31;
  const int wid  = tid >> 5;
  const int mw   = wid & 3;          // which 16-row query tile
  const int ng   = wid >> 2;         // which 64-key group (0/1)
  const int half = lane >> 4;
  const int l15  = lane & 15;

  const int bbase      = blockIdx.x * TB;
  const int chunk_base = blockIdx.y * CHUNK;

  float mins[8];
  int   idxs[8];
#pragma unroll
  for (int r = 0; r < 8; ++r) { mins[r] = 3.4e38f; idxs[r] = 0; }

  // per-thread staging slots (A: 2 float4 of 64x32, B: 4 float4 of 128x32)
  const int arow0 = (tid * 2 + 0) >> 3, ac0 = ((tid * 2 + 0) & 7) * 4;
  const int arow1 = (tid * 2 + 1) >> 3, ac1 = ((tid * 2 + 1) & 7) * 4;
  int brow[4], bc[4];
#pragma unroll
  for (int u = 0; u < 4; ++u) {
    int slot = tid + 256 * u;
    brow[u] = slot >> 3;
    bc[u]   = (slot & 7) * 4;
  }

  for (int pass = 0; pass < PASSES; ++pass) {
    const int nbase = chunk_base + pass * TN;
    v8f acc[4];
#pragma unroll
    for (int t = 0; t < 4; ++t) {
      v8f z = {0.f, 0.f, 0.f, 0.f, 0.f, 0.f, 0.f, 0.f};
      acc[t] = z;
    }

    // prefetch slice 0 into registers
    float4 a4[2], b4[4];
    {
      const int kk0 = 0;  // slice 0 always lies in X (DXX = 24*KS)
      a4[0] = *(const float4*)(X + (size_t)(bbase + arow0) * DXX + kk0 + ac0);
      a4[1] = *(const float4*)(X + (size_t)(bbase + arow1) * DXX + kk0 + ac1);
#pragma unroll
      for (int u = 0; u < 4; ++u)
        b4[u] = *(const float4*)(keys + (size_t)(nbase + brow[u]) * D_TOT + kk0 + bc[u]);
    }

    for (int ks = 0; ks < KSTEPS; ++ks) {
      __syncthreads();  // previous compute finished reading LDS
      // ---- convert (hi/lo split) once per WG and store packed pairs ------
      {
        uint4 w0 = cvt_hilo(a4[0]);
        uint4 w1 = cvt_hilo(a4[1]);
        *(uint2*)&As_hi[arow0][ac0 >> 1] = make_uint2(w0.x, w0.y);
        *(uint2*)&As_lo[arow0][ac0 >> 1] = make_uint2(w0.z, w0.w);
        *(uint2*)&As_hi[arow1][ac1 >> 1] = make_uint2(w1.x, w1.y);
        *(uint2*)&As_lo[arow1][ac1 >> 1] = make_uint2(w1.z, w1.w);
#pragma unroll
        for (int u = 0; u < 4; ++u) {
          uint4 w = cvt_hilo(b4[u]);
          *(uint2*)&Bs_hi[brow[u]][bc[u] >> 1] = make_uint2(w.x, w.y);
          *(uint2*)&Bs_lo[brow[u]][bc[u] >> 1] = make_uint2(w.z, w.w);
        }
      }
      __syncthreads();  // staged data visible

      // ---- issue next slice's global loads (hide latency behind WMMAs) ---
      if (ks + 1 < KSTEPS) {
        const int kk0 = (ks + 1) * KS;
        const float* s0 = (kk0 < DXX) ? (X + (size_t)(bbase + arow0) * DXX + kk0 + ac0)
                                      : (Y + (size_t)(bbase + arow0) * DYY + (kk0 - DXX) + ac0);
        const float* s1 = (kk0 < DXX) ? (X + (size_t)(bbase + arow1) * DXX + kk0 + ac1)
                                      : (Y + (size_t)(bbase + arow1) * DYY + (kk0 - DXX) + ac1);
        a4[0] = *(const float4*)s0;
        a4[1] = *(const float4*)s1;
#pragma unroll
        for (int u = 0; u < 4; ++u)
          b4[u] = *(const float4*)(keys + (size_t)(nbase + brow[u]) * D_TOT + kk0 + bc[u]);
      }

      // ---- build fragments with pure ds_load_b128 (no VALU) --------------
      // A 16-bit 16x32 layout: VGPR j holds K pairs p = (j>=4 ? 8:0) + half*4 + (j&3)
      FragCast Ahi, Alo;
      {
        const unsigned* ph = &As_hi[mw * 16 + l15][0];
        const unsigned* pl = &As_lo[mw * 16 + l15][0];
        uint4 h0 = *(const uint4*)(ph + (half << 2));
        uint4 h1 = *(const uint4*)(ph + 8 + (half << 2));
        uint4 g0 = *(const uint4*)(pl + (half << 2));
        uint4 g1 = *(const uint4*)(pl + 8 + (half << 2));
        v8u uh = {h0.x, h0.y, h0.z, h0.w, h1.x, h1.y, h1.z, h1.w};
        v8u ul = {g0.x, g0.y, g0.z, g0.w, g1.x, g1.y, g1.z, g1.w};
        Ahi.u = uh; Alo.u = ul;
      }
#pragma unroll
      for (int t = 0; t < 4; ++t) {
        // B 32x16 layout: lane n = l15; K pairs p = (j>=4 ? 8:0) + half*4 + (j&3)
        const int nr = ng * 64 + t * 16 + l15;
        const unsigned* ph = &Bs_hi[nr][0];
        const unsigned* pl = &Bs_lo[nr][0];
        uint4 h0 = *(const uint4*)(ph + (half << 2));
        uint4 h1 = *(const uint4*)(ph + 8 + (half << 2));
        uint4 g0 = *(const uint4*)(pl + (half << 2));
        uint4 g1 = *(const uint4*)(pl + 8 + (half << 2));
        FragCast Bhi, Blo;
        v8u uh = {h0.x, h0.y, h0.z, h0.w, h1.x, h1.y, h1.z, h1.w};
        v8u ul = {g0.x, g0.y, g0.z, g0.w, g1.x, g1.y, g1.z, g1.w};
        Bhi.u = uh; Blo.u = ul;

        acc[t] = __builtin_amdgcn_wmma_f32_16x16x32_bf16(
            false, Ahi.b, false, Bhi.b, (short)0, acc[t], false, false);
        acc[t] = __builtin_amdgcn_wmma_f32_16x16x32_bf16(
            false, Ahi.b, false, Blo.b, (short)0, acc[t], false, false);
        acc[t] = __builtin_amdgcn_wmma_f32_16x16x32_bf16(
            false, Alo.b, false, Bhi.b, (short)0, acc[t], false, false);
      }
    }

    // Fold this pass's 64 key columns into the running per-lane (min, idx).
    // C layout: VGPR r, lane -> M = r + 8*half, N = lane&15.
#pragma unroll
    for (int t = 0; t < 4; ++t) {
      int   n  = nbase + ng * 64 + t * 16 + l15;
      float kn = knorm[n];
#pragma unroll
      for (int r = 0; r < 8; ++r) {
        float s = kn - 2.0f * acc[t][r];
        if (s < mins[r] || (s == mins[r] && n < idxs[r])) {
          mins[r] = s;
          idxs[r] = n;
        }
      }
    }
  }

  // Reduce across the 16 N-lanes of each half; tie-break on lowest index.
#pragma unroll
  for (int r = 0; r < 8; ++r) {
    float mv = mins[r];
    int   mi = idxs[r];
#pragma unroll
    for (int off = 1; off < 16; off <<= 1) {
      float om = __shfl_xor(mv, off, 16);
      int   oi = __shfl_xor(mi, off, 16);
      if (om < mv || (om == mv && oi < mi)) { mv = om; mi = oi; }
    }
    if (l15 == 0) {
      int b    = bbase + mw * 16 + (half << 3) + r;
      int part = blockIdx.y * 2 + ng;
      pmin[(size_t)b * PARTS + part] = mv;
      pidx[(size_t)b * PARTS + part] = mi;
    }
  }
}

// ------ Kernel 3: reduce partials, q-norm, threshold, permuted gather ------
__global__ __launch_bounds__(256) void finalize_kernel(
    const float* __restrict__ values, const int* __restrict__ pi,
    const float* __restrict__ X, const float* __restrict__ Y,
    const float* __restrict__ pmin, const int* __restrict__ pidx,
    float* __restrict__ out) {
  const int b = blockIdx.x;
  const int t = threadIdx.x;
  __shared__ float sm[256];
  __shared__ int   si[256];
  __shared__ float sq[256];

  float mv = 3.4e38f;
  int   mi = 0;
  for (int p = t; p < PARTS; p += 256) {
    float v = pmin[(size_t)b * PARTS + p];
    int   i = pidx[(size_t)b * PARTS + p];
    if (v < mv || (v == mv && i < mi)) { mv = v; mi = i; }
  }
  sm[t] = mv;
  si[t] = mi;

  float qn = 0.f;
  for (int d = t; d < DXX; d += 256) { float x = X[(size_t)b * DXX + d]; qn += x * x; }
  for (int d = t; d < DYY; d += 256) { float y = Y[(size_t)b * DYY + d]; qn += y * y; }
  sq[t] = qn;
  __syncthreads();

  for (int s = 128; s > 0; s >>= 1) {
    if (t < s) {
      float ov = sm[t + s];
      int   oi = si[t + s];
      if (ov < sm[t] || (ov == sm[t] && oi < si[t])) { sm[t] = ov; si[t] = oi; }
      sq[t] += sq[t + s];
    }
    __syncthreads();
  }

  const float dmin  = sm[0] + sq[0];          // min ||k - q||^2
  const float scale = (dmin <= THRESHV) ? 1.f : 0.f;
  const int   row   = pi[si[0]];
  for (int d = t; d < DXX; d += 256)
    out[(size_t)b * DXX + d] = scale * values[(size_t)row * DXX + d];
}

extern "C" void kernel_launch(void* const* d_in, const int* in_sizes, int n_in,
                              void* d_out, int out_size, void* d_ws, size_t ws_size,
                              hipStream_t stream) {
  const float* keys   = (const float*)d_in[0];
  const float* values = (const float*)d_in[1];
  const int*   pi     = (const int*)d_in[2];
  const float* X      = (const float*)d_in[3];
  const float* Y      = (const float*)d_in[4];
  float*       out    = (float*)d_out;

  // workspace layout: pmin (BQ*PARTS f32) | pidx (BQ*PARTS i32) | knorm (N f32)
  float* pmin  = (float*)d_ws;
  int*   pidx  = (int*)(pmin + (size_t)BQ * PARTS);
  float* knorm = (float*)(pidx + (size_t)BQ * PARTS);

  knorm_kernel<<<N_KEYS / 256, 256, 0, stream>>>(keys, knorm);

  dim3 grid(BQ / TB, NSPLIT);   // x = query tiles (L2 reuse of key chunks across x)
  dist_argmin_kernel<<<grid, 256, 0, stream>>>(keys, knorm, X, Y, pmin, pidx);

  finalize_kernel<<<BQ, 256, 0, stream>>>(values, pi, X, Y, pmin, pidx, out);
}
